// MIEstimator_12360915878462
// MI455X (gfx1250) — matrix operand order repaired and checked by reading, not compiled, not run on hardware
//
#include <hip/hip_runtime.h>
#include <hip/hip_bf16.h>

typedef __attribute__((ext_vector_type(16))) __bf16 v16bf;
typedef __attribute__((ext_vector_type(8)))  float  v8f;
typedef unsigned short u16;
typedef unsigned int   u32;

// ---------------- helpers ----------------

// f32 -> bf16, round-to-nearest-even
__device__ __forceinline__ u16 f2bf(float f) {
    u32 u = __float_as_uint(f);
    u += 0x7FFFu + ((u >> 16) & 1u);
    return (u16)(u >> 16);
}

union ABfrag { v16bf v; uint4 q[2]; };
union CT     { v8f   v; float f[8]; };

// A-fragment (16x32 bf16, MxK) from an LDS row (row stride 256 halves).
// Layout used here (contiguous k, same as B): lane M = lane&15; lane group g=lane>>4 holds halves
//   i=0..15 -> K = kbase + 16g + i
__device__ __forceinline__ v16bf load_a_lds(const u16* rowp, int kbase, int g) {
    ABfrag u;
    u.q[0] = *(const uint4*)(rowp + kbase + 16 * g);
    u.q[1] = *(const uint4*)(rowp + kbase + 16 * g + 8);
    return u.v;
}

// B-fragment (32x16 bf16, KxN) from transposed bf16 weights wT[n][k].
// Layout used here (contiguous k, same as A): lane N = lane&15; lane group g holds K = kbase + 16g + i
// -> 16 contiguous halves starting at wT + n*ldk + kbase + 16g
__device__ __forceinline__ v16bf load_b_glb(const u16* p) {
    ABfrag u;
    u.q[0] = *(const uint4*)(p);
    u.q[1] = *(const uint4*)(p + 8);
    return u.v;
}

// gfx1250 async DMA: global (GVS mode: SGPR base + per-lane 32-bit offset)
// -> LDS (per-lane LDS byte address in VDST VGPR). Tracked by ASYNCcnt.
__device__ __forceinline__ void async_load_b128(unsigned lds_addr,
                                                const void* gbase,
                                                unsigned goff) {
    asm volatile("global_load_async_to_lds_b128 %0, %1, %2"
                 :
                 : "v"(lds_addr), "v"(goff), "s"(gbase)
                 : "memory");
}

__device__ __forceinline__ void wait_asynccnt0() {
    asm volatile("s_wait_asynccnt 0x0" ::: "memory");
}

// Low 32 bits of a generic pointer to __shared__ data == LDS byte address
// (ISA flat-aperture rule: LDS address is the low 32 bits).
__device__ __forceinline__ unsigned lds_addr_of(const void* p) {
    return (unsigned)(uintptr_t)p;
}

// ---------------- kernels ----------------

__global__ void zero_out_kernel(float* out) {
    if (threadIdx.x == 0 && blockIdx.x == 0) out[0] = 0.0f;
}

// Per-feature mean(y), mean(y^2) over N=8192. One block per column n (DY=64).
__global__ __launch_bounds__(256) void ystats_kernel(const float* __restrict__ y,
                                                     float* __restrict__ ybar,
                                                     float* __restrict__ y2bar) {
    __shared__ float s1[256];
    __shared__ float s2[256];
    const int n = blockIdx.x;
    float a = 0.0f, b = 0.0f;
    for (int r = threadIdx.x; r < 8192; r += 256) {
        float v = y[r * 64 + n];
        a += v;
        b += v * v;
    }
    s1[threadIdx.x] = a;
    s2[threadIdx.x] = b;
    __syncthreads();
    for (int off = 128; off > 0; off >>= 1) {
        if (threadIdx.x < off) {
            s1[threadIdx.x] += s1[threadIdx.x + off];
            s2[threadIdx.x] += s2[threadIdx.x + off];
        }
        __syncthreads();
    }
    if (threadIdx.x == 0) {
        ybar[n]  = s1[0] * (1.0f / 8192.0f);
        y2bar[n] = s2[0] * (1.0f / 8192.0f);
    }
}

// Transpose + convert weights to bf16: wT[n*ldk + k] = bf16(w[k*ldn + n]).
__global__ __launch_bounds__(256) void prep_weights_kernel(
        const float* __restrict__ w1mu, const float* __restrict__ w1lv,
        const float* __restrict__ w2mu, const float* __restrict__ w2lv,
        u16* __restrict__ w1Tmu, u16* __restrict__ w1Tlv,
        u16* __restrict__ w2Tmu, u16* __restrict__ w2Tlv) {
    const int idx = blockIdx.x * 256 + threadIdx.x;   // 65536 threads
    const int n = idx >> 8;
    const int k = idx & 255;
    w1Tmu[idx] = f2bf(w1mu[k * 256 + n]);
    w1Tlv[idx] = f2bf(w1lv[k * 256 + n]);
    if (idx < 64 * 256) {
        w2Tmu[idx] = f2bf(w2mu[k * 64 + n]);
        w2Tlv[idx] = f2bf(w2lv[k * 64 + n]);
    }
}

// One MLP head for a 16-row tile: out = (relu(X@W1 + b1)) @ W2 + b2.
__device__ __forceinline__ void mlp_head(const u16* __restrict__ xw,
                                         u16* __restrict__ hw,
                                         const u16* __restrict__ w1T,
                                         const float* __restrict__ b1,
                                         const u16* __restrict__ w2T,
                                         const float* __restrict__ b2,
                                         int lane, float (*out)[8]) {
    const int g  = lane >> 4;
    const int ln = lane & 15;

    // ---- GEMM1: 16x256 @ 256x256, K in 8 steps of 32 ----
    v16bf a[8];
#pragma unroll
    for (int k = 0; k < 8; ++k) a[k] = load_a_lds(xw + ln * 256, 32 * k, g);

    for (int nt = 0; nt < 16; ++nt) {
        CT c;
        c.v = (v8f){0.f, 0.f, 0.f, 0.f, 0.f, 0.f, 0.f, 0.f};
        const u16* wp = w1T + (nt * 16 + ln) * 256 + 16 * g;
#pragma unroll
        for (int k = 0; k < 8; ++k) {
            v16bf b = load_b_glb(wp + 32 * k);
            c.v = __builtin_amdgcn_wmma_f32_16x16x32_bf16(
                false, a[k], false, b, (short)0, c.v, false, false);
        }
        // bias + ReLU, store hidden tile to LDS as bf16.
        // C layout: VGPR r holds (M = r + 8g, N = lane&15).
        const float bb = b1[nt * 16 + ln];
#pragma unroll
        for (int r = 0; r < 8; ++r) {
            float v = c.f[r] + bb;
            v = v > 0.0f ? v : 0.0f;
            hw[(r + 8 * g) * 256 + nt * 16 + ln] = f2bf(v);
        }
    }
    __syncthreads();   // hidden fully resident in LDS (uniform across block)

    // ---- GEMM2: 16x256 @ 256x64 ----
    v16bf a2[8];
#pragma unroll
    for (int k = 0; k < 8; ++k) a2[k] = load_a_lds(hw + ln * 256, 32 * k, g);

#pragma unroll
    for (int nt = 0; nt < 4; ++nt) {
        CT c;
        c.v = (v8f){0.f, 0.f, 0.f, 0.f, 0.f, 0.f, 0.f, 0.f};
        const u16* wp = w2T + (nt * 16 + ln) * 256 + 16 * g;
#pragma unroll
        for (int k = 0; k < 8; ++k) {
            v16bf b = load_b_glb(wp + 32 * k);
            c.v = __builtin_amdgcn_wmma_f32_16x16x32_bf16(
                false, a2[k], false, b, (short)0, c.v, false, false);
        }
        const float bb = b2[nt * 16 + ln];
#pragma unroll
        for (int r = 0; r < 8; ++r) out[nt][r] = c.f[r] + bb;
    }
    __syncthreads();   // allow hw reuse by the next head
}

// Fused: both heads + closed-form loss + global reduction.
// 1 wave = 16 rows; 4 waves/block; 128 blocks cover N = 8192.
__global__ __launch_bounds__(128) void mi_main_kernel(
        const float* __restrict__ x, const float* __restrict__ y,
        const float* __restrict__ mub1, const float* __restrict__ mub2,
        const float* __restrict__ lvb1, const float* __restrict__ lvb2,
        const u16* __restrict__ w1Tmu, const u16* __restrict__ w1Tlv,
        const u16* __restrict__ w2Tmu, const u16* __restrict__ w2Tlv,
        const float* __restrict__ ybar, const float* __restrict__ y2bar,
        float* __restrict__ out) {
    // 16KB/wave f32 staging, later reused as the bf16 hidden buffer.
    __shared__ __align__(16) float stagef[4][16 * 256];
    __shared__ __align__(16) u16   xt[4][16 * 256];   // bf16 input tiles, 8KB/wave
    __shared__ float red[4];

    const int lane = threadIdx.x & 31;
    const int wid  = threadIdx.x >> 5;
    const int g    = lane >> 4;
    const int ln   = lane & 15;
    const int m0   = (blockIdx.x * 4 + wid) * 16;   // first row of this wave's tile

    float* sf = stagef[wid];
    u16*   hw = (u16*)sf;      // hidden buffer aliases dead f32 staging
    u16*   xw = xt[wid];

    // Warm GL2 for the weight streams (global_prefetch_b8).
    if (threadIdx.x == 0) {
        __builtin_prefetch(w1Tlv, 0, 0);
        __builtin_prefetch(w1Tmu, 0, 0);
        __builtin_prefetch(w2Tlv, 0, 0);
        __builtin_prefetch(w2Tmu, 0, 0);
    }

    // 1) Async DMA the raw f32 x tile (rows m0..m0+15 are 16KB contiguous)
    //    straight into LDS: 1024 x 16B per wave, tracked by ASYNCcnt.
    {
        const unsigned lds0 = lds_addr_of(sf);
        const void* gx = (const void*)(x + m0 * 256);
        for (int it = lane; it < 1024; it += 32) {
            const unsigned off = (unsigned)it * 16u;
            async_load_b128(lds0 + off, gx, off);
        }
        wait_asynccnt0();
    }

    // 2) Convert staged f32 -> bf16 tile (LDS -> LDS).
    for (int it = lane; it < 1024; it += 32) {
        const float4 f = *(const float4*)(sf + it * 4);
        ushort4 h;
        h.x = f2bf(f.x); h.y = f2bf(f.y); h.z = f2bf(f.z); h.w = f2bf(f.w);
        *(ushort4*)(xw + it * 4) = h;
    }
    __syncthreads();   // also fences f32 reads vs. later u16 hidden writes

    // logvar head first -> inv_var kept in registers
    float lvr[4][8];
    mlp_head(xw, hw, w1Tlv, lvb1, w2Tlv, lvb2, lane, lvr);
    float iv[4][8];
#pragma unroll
    for (int t = 0; t < 4; ++t)
#pragma unroll
        for (int r = 0; r < 8; ++r) iv[t][r] = __expf(-tanhf(lvr[t][r]));

    // mu head
    float mu[4][8];
    mlp_head(xw, hw, w1Tmu, mub1, w2Tmu, mub2, lane, mu);

    // Closed form: pos - neg = -0.5*iv*((y^2 - y2bar) - 2*mu*(y - ybar))
    float s = 0.0f;
#pragma unroll
    for (int t = 0; t < 4; ++t) {
        const int n     = t * 16 + ln;
        const float yb  = ybar[n];
        const float y2b = y2bar[n];
#pragma unroll
        for (int r = 0; r < 8; ++r) {
            const int row  = m0 + r + 8 * g;
            const float yy = y[row * 64 + n];
            s += -0.5f * iv[t][r] * ((yy * yy - y2b) - 2.0f * mu[t][r] * (yy - yb));
        }
    }

    // wave32 shuffle reduce, then block reduce, then one atomic (scaled by 1/N)
#pragma unroll
    for (int off = 16; off > 0; off >>= 1) s += __shfl_down(s, off, 32);
    if (lane == 0) red[wid] = s;
    __syncthreads();
    if (threadIdx.x == 0) {
        const float tot = red[0] + red[1] + red[2] + red[3];
        atomicAdd(out, tot * (1.0f / 8192.0f));
    }
}

// ---------------- launch ----------------

extern "C" void kernel_launch(void* const* d_in, const int* in_sizes, int n_in,
                              void* d_out, int out_size, void* d_ws, size_t ws_size,
                              hipStream_t stream) {
    const float* x    = (const float*)d_in[0];   // emb_x  (8192,256)
    const float* y    = (const float*)d_in[1];   // emb_y  (8192,64)
    const float* muw1 = (const float*)d_in[2];   // (256,256)
    const float* mub1 = (const float*)d_in[3];   // (256,)
    const float* muw2 = (const float*)d_in[4];   // (256,64)
    const float* mub2 = (const float*)d_in[5];   // (64,)
    const float* lvw1 = (const float*)d_in[6];
    const float* lvb1 = (const float*)d_in[7];
    const float* lvw2 = (const float*)d_in[8];
    const float* lvb2 = (const float*)d_in[9];

    char* ws = (char*)d_ws;
    u16* w1Tmu = (u16*)(ws);                 // 256*256*2 = 131072 B
    u16* w1Tlv = (u16*)(ws + 131072);        // 131072 B
    u16* w2Tmu = (u16*)(ws + 262144);        // 64*256*2 = 32768 B
    u16* w2Tlv = (u16*)(ws + 294912);        // 32768 B
    float* ybar  = (float*)(ws + 327680);    // 256 B
    float* y2bar = (float*)(ws + 327936);    // 256 B

    float* out = (float*)d_out;

    zero_out_kernel<<<1, 64, 0, stream>>>(out);
    ystats_kernel<<<64, 256, 0, stream>>>(y, ybar, y2bar);
    prep_weights_kernel<<<256, 256, 0, stream>>>(muw1, lvw1, muw2, lvw2,
                                                 w1Tmu, w1Tlv, w2Tmu, w2Tlv);
    mi_main_kernel<<<128, 128, 0, stream>>>(x, y, mub1, mub2, lvb1, lvb2,
                                            w1Tmu, w1Tlv, w2Tmu, w2Tlv,
                                            ybar, y2bar, out);
}
